// SwinTransFormer_2456721293549
// MI455X (gfx1250) — matrix operand order repaired and checked by reading, not compiled, not run on hardware
//
#include <hip/hip_runtime.h>

typedef __attribute__((ext_vector_type(16))) _Float16 v16h;
typedef __attribute__((ext_vector_type(8)))  _Float16 h8v;
typedef __attribute__((ext_vector_type(4)))  _Float16 h4;
typedef __attribute__((ext_vector_type(8)))  float    v8f;

#define NH 6
#define HD 32
#define HW 65536            // 256*256
#define SCALE 0.17677669529663687f

#define ST_QK 40            // Q/K row-major [token][d] stride (halves), rows 16B aligned
#define ST_V  40            // Vt [d][token] stride (halves)
#define ST_P  72            // P row-major stride (halves), rows 16B aligned
#define ST_S  68            // S/Ot stride (f32)

// per-wave shared memory layout (bytes); buffers overlaid across phases
#define OFF_Q 0             // Q : 64 x 40 f16 = 5120   (Q pre-scaled by SCALE)
#define OFF_K 5120          // K : 64 x 40 f16 = 5120
#define OFF_P 0             // P : 64 x 72 f16 = 9216   (overlays Q+K, dead after QK)
#define OFF_V 10240         // Vt: 32 x 40 f16 = 2560
#define OFF_S 12800         // S : 64 x 68 f32 = 17408
#define OFF_O 12800         // Ot: 32 x 68 f32 = 8704   (overlays S, dead after softmax)
#define OFF_R 30208         // rpbE[qj][g][kj]: 8*15*8 f32 = 3840
#define WAVE_SMEM 34048

__device__ __forceinline__ v16h cat8(h8v lo, h8v hi) {
    return __builtin_shufflevector(lo, hi, 0,1,2,3,4,5,6,7,8,9,10,11,12,13,14,15);
}

__global__ __launch_bounds__(32) void swin_attn_kernel(
    const float* __restrict__ x, const float* __restrict__ rpb_table,
    float* __restrict__ out)
{
    __shared__ __align__(16) char sm[WAVE_SMEM];
    _Float16* Qr   = (_Float16*)(sm + OFF_Q);
    _Float16* Kr   = (_Float16*)(sm + OFF_K);
    _Float16* Vt   = (_Float16*)(sm + OFF_V);
    _Float16* Pm   = (_Float16*)(sm + OFF_P);
    float*    S    = (float*)(sm + OFF_S);
    float*    Ot   = (float*)(sm + OFF_O);
    float*    rpbE = (float*)(sm + OFF_R);

    const int lane = threadIdx.x & 31;

    // wave -> (head, b, wy, wx); wx innermost for cross-wave coalescing
    int wg = blockIdx.x;                    // 0..24575
    const int wx   = wg & 31;  wg >>= 5;
    const int wy   = wg & 31;  wg >>= 5;
    const int b    = wg & 3;   wg >>= 2;
    const int head = wg;                    // 0..5
    const bool needMask = (wy == 31) || (wx == 31);

    // expand rpb for this head: rpbE[qj][g][kj] = rpb[g*15 + 7 + qj - kj]
    // -> per (row, key-group) bias is one aligned 32B run (2 x b128)
    for (int e = lane; e < 960; e += 32) {
        int qj = e / 120, g = (e >> 3) % 15, kj = e & 7;
        rpbE[e] = rpb_table[(g * 15 + 7 + qj - kj) * NH + head];
    }

    const int hi  = lane >> 4;              // half-wave select
    const int l15 = lane & 15;

    // pixel-quad mapping for global <-> LDS staging (float4, never wraps)
    const int pi  = l15 >> 1;               // window row 0..7
    const int pj0 = (l15 & 1) * 4;          // window col 0 or 4
    const int gh  = (wy * 8 + pi + 4) & 255;    // shift folded into address
    const int gw  = (wx * 8 + pj0 + 4) & 255;
    const long spat = (long)gh * 256 + gw;
    const int tok0 = pi * 8 + pj0;

    // ---- stage Q (pre-scaled), K row-major [token][d]: 4 d-planes/lane, b64 stores ----
    #pragma unroll
    for (int t = 0; t < 2; ++t) {
        _Float16* dst = t ? Kr : Qr;
        const float scl = t ? 1.0f : SCALE;
        const float* src = x + ((((long)b * 3 + t) * NH + head) * HD) * HW + spat;
        #pragma unroll
        for (int g = 0; g < 4; ++g) {
            int d0 = g * 8 + hi * 4;
            float4 v0 = *(const float4*)(src + (long)(d0 + 0) * HW);
            float4 v1 = *(const float4*)(src + (long)(d0 + 1) * HW);
            float4 v2 = *(const float4*)(src + (long)(d0 + 2) * HW);
            float4 v3 = *(const float4*)(src + (long)(d0 + 3) * HW);
            h4 s0 = {(_Float16)(v0.x*scl), (_Float16)(v1.x*scl), (_Float16)(v2.x*scl), (_Float16)(v3.x*scl)};
            h4 s1 = {(_Float16)(v0.y*scl), (_Float16)(v1.y*scl), (_Float16)(v2.y*scl), (_Float16)(v3.y*scl)};
            h4 s2 = {(_Float16)(v0.z*scl), (_Float16)(v1.z*scl), (_Float16)(v2.z*scl), (_Float16)(v3.z*scl)};
            h4 s3 = {(_Float16)(v0.w*scl), (_Float16)(v1.w*scl), (_Float16)(v2.w*scl), (_Float16)(v3.w*scl)};
            *(h4*)(dst + (tok0 + 0) * ST_QK + d0) = s0;
            *(h4*)(dst + (tok0 + 1) * ST_QK + d0) = s1;
            *(h4*)(dst + (tok0 + 2) * ST_QK + d0) = s2;
            *(h4*)(dst + (tok0 + 3) * ST_QK + d0) = s3;
        }
    }
    // ---- stage V transposed [d][token] ----
    #pragma unroll
    for (int dd = 0; dd < 16; ++dd) {
        int d = dd * 2 + hi;
        const float4 v = *(const float4*)(x +
            ((((long)b * 3 + 2) * NH + head) * HD + d) * HW + spat);
        h4 hv = { (_Float16)v.x, (_Float16)v.y, (_Float16)v.z, (_Float16)v.w };
        *(h4*)(Vt + d * ST_V + tok0) = hv;
    }
    __syncthreads();

    // ---- S = (Q*scale) K^T : pure WMMA + stores, no interleaved loads ----
    v16h Bq[4];
    #pragma unroll
    for (int nt = 0; nt < 4; ++nt) {
        const _Float16* kp = Kr + (nt * 16 + l15) * ST_QK + hi * 8;
        Bq[nt] = cat8(*(const h8v*)kp, *(const h8v*)(kp + 16));
    }
    #pragma unroll
    for (int mt = 0; mt < 4; ++mt) {
        const int m0 = mt * 16;
        const _Float16* qp = Qr + (m0 + l15) * ST_QK + hi * 8;
        v16h A = cat8(*(const h8v*)qp, *(const h8v*)(qp + 16));
        v8f acc[4];
        #pragma unroll
        for (int nt = 0; nt < 4; ++nt)
            acc[nt] = __builtin_amdgcn_wmma_f32_16x16x32_f16(
                false, A, false, Bq[nt], (short)0, v8f{}, false, false);
        #pragma unroll
        for (int nt = 0; nt < 4; ++nt)
            #pragma unroll
            for (int r = 0; r < 8; ++r)
                S[(m0 + 8 * hi + r) * ST_S + nt * 16 + l15] = acc[nt][r];
    }
    __syncthreads();

    // ---- row softmax with fused rpb + shift-mask (row register-resident) ----
    #pragma unroll
    for (int rr = 0; rr < 2; ++rr) {
        const int m = lane + rr * 32;
        const int qi = m >> 3, qj = m & 7;
        float4 row[16];
        float* rowf = (float*)row;
        #pragma unroll
        for (int c = 0; c < 16; ++c) row[c] = *(const float4*)(S + m * ST_S + 4 * c);
        #pragma unroll
        for (int ki = 0; ki < 8; ++ki) {        // bias: 2 x b128 per key-group
            const float* bp = rpbE + (qj * 15 + (qi - ki + 7)) * 8;
            float4 b0 = *(const float4*)bp, b1 = *(const float4*)(bp + 4);
            row[2*ki].x   += b0.x; row[2*ki].y   += b0.y;
            row[2*ki].z   += b0.z; row[2*ki].w   += b0.w;
            row[2*ki+1].x += b1.x; row[2*ki+1].y += b1.y;
            row[2*ki+1].z += b1.z; row[2*ki+1].w += b1.w;
        }
        if (needMask) {                         // only boundary windows (6% of waves)
            const int qgh = wy * 8 + qi, qgw = wx * 8 + qj;
            const int cq = 3 * ((qgh >= 248) + (qgh >= 252)) + (qgw >= 248) + (qgw >= 252);
            #pragma unroll
            for (int ki = 0; ki < 8; ++ki) {
                const int kgh = wy * 8 + ki;
                const int c3 = 3 * ((kgh >= 248) + (kgh >= 252));   // uniform
                #pragma unroll
                for (int kj = 0; kj < 8; ++kj) {
                    const int kgw = wx * 8 + kj;
                    const int ckv = c3 + (kgw >= 248) + (kgw >= 252); // uniform
                    rowf[ki * 8 + kj] += (ckv != cq) ? -100.0f : 0.0f;
                }
            }
        }
        float mx = -3.0e38f;
        #pragma unroll
        for (int c = 0; c < 16; ++c)
            mx = fmaxf(mx, fmaxf(fmaxf(row[c].x, row[c].y), fmaxf(row[c].z, row[c].w)));
        float sum = 0.f;
        #pragma unroll
        for (int c = 0; c < 16; ++c) {
            row[c].x = __expf(row[c].x - mx);
            row[c].y = __expf(row[c].y - mx);
            row[c].z = __expf(row[c].z - mx);
            row[c].w = __expf(row[c].w - mx);
            sum += (row[c].x + row[c].y) + (row[c].z + row[c].w);
        }
        float inv = 1.0f / sum;
        #pragma unroll
        for (int c = 0; c < 8; ++c) {
            h8v p = { (_Float16)(row[2*c].x * inv),   (_Float16)(row[2*c].y * inv),
                      (_Float16)(row[2*c].z * inv),   (_Float16)(row[2*c].w * inv),
                      (_Float16)(row[2*c+1].x * inv), (_Float16)(row[2*c+1].y * inv),
                      (_Float16)(row[2*c+1].z * inv), (_Float16)(row[2*c+1].w * inv) };
            *(h8v*)(Pm + m * ST_P + 8 * c) = p;
        }
    }
    __syncthreads();

    // ---- O = P V : 8 tiles x 2 k-steps of v_wmma_f32_16x16x32_f16 ----
    v16h Bf[2][2];
    #pragma unroll
    for (int nt = 0; nt < 2; ++nt)
        #pragma unroll
        for (int kc = 0; kc < 2; ++kc) {
            const _Float16* vp = Vt + (nt * 16 + l15) * ST_V + kc * 32 + hi * 8;
            Bf[nt][kc] = cat8(*(const h8v*)vp, *(const h8v*)(vp + 16));
        }
    #pragma unroll
    for (int mt = 0; mt < 4; ++mt) {
        const int m0 = mt * 16;
        const _Float16* ap = Pm + (m0 + l15) * ST_P + hi * 8;
        v16h A0 = cat8(*(const h8v*)ap,        *(const h8v*)(ap + 16));
        v16h A1 = cat8(*(const h8v*)(ap + 32), *(const h8v*)(ap + 48));
        #pragma unroll
        for (int nt = 0; nt < 2; ++nt) {
            v8f acc = __builtin_amdgcn_wmma_f32_16x16x32_f16(
                false, A0, false, Bf[nt][0], (short)0, v8f{}, false, false);
            acc = __builtin_amdgcn_wmma_f32_16x16x32_f16(
                false, A1, false, Bf[nt][1], (short)0, acc, false, false);
            #pragma unroll
            for (int r = 0; r < 8; ++r)
                Ot[(nt * 16 + l15) * ST_S + m0 + r + 8 * hi] = acc[r];
        }
    }
    __syncthreads();

    // ---- store: LDS f32 -> global float4 (un-shift folded into address) ----
    #pragma unroll
    for (int dd = 0; dd < 16; ++dd) {
        int d = dd * 2 + hi;
        float4 o = *(const float4*)(Ot + d * ST_S + tok0);
        *(float4*)(out + (((long)b * NH + head) * HD + d) * HW + spat) = o;
    }
}

extern "C" void kernel_launch(void* const* d_in, const int* in_sizes, int n_in,
                              void* d_out, int out_size, void* d_ws, size_t ws_size,
                              hipStream_t stream) {
    const float* x   = (const float*)d_in[0];   // (4, 3*192, 256, 256) f32
    const float* rpb = (const float*)d_in[1];   // (225, 6) f32
    float* out = (float*)d_out;                 // (4, 192, 256, 256) f32
    (void)in_sizes; (void)n_in; (void)out_size; (void)d_ws; (void)ws_size;
    // one wave32 per (window, head)
    swin_attn_kernel<<<dim3(24576), dim3(32), 0, stream>>>(x, rpb, out);
}
